// PointNetFeaturePropagation_27504970563939
// MI455X (gfx1250) — matrix operand order repaired and checked by reading, not compiled, not run on hardware
//
#include <hip/hip_runtime.h>

// ---------------------------------------------------------------------------
// PointNet feature propagation, fused for MI455X (gfx1250, wave32, WMMA).
//   k1: top-3 NN + inverse-distance weights (fp32 VALU, centres in LDS)
//   k2: weight transpose->fp16 K-major + BN fold
//   k3: gather/concat -> GEMM1(320x256) -> BN/ReLU -> GEMM2(256x128) -> BN/ReLU
//       v_wmma_f32_16x16x32_f16; B-fragments register-resident per wave
//       (each block reads W0t/W1t exactly once), A-fragments streamed from LDS.
// ---------------------------------------------------------------------------

#define N_PTS 65536
#define S_PTS 2048

typedef __attribute__((ext_vector_type(16))) _Float16 v16h;
typedef __attribute__((ext_vector_type(8)))  float    v8f;
typedef __attribute__((ext_vector_type(4)))  float    f32x4;

union FragH { v16h v; f32x4 q[2]; };

// ---------------------------------------------------------------- kernel 1 --
__global__ __launch_bounds__(256) void top3_kernel(
    const float* __restrict__ xyz1, const float* __restrict__ xyz2,
    int* __restrict__ idx, float* __restrict__ wgt) {
  __shared__ float s2x[S_PTS], s2y[S_PTS], s2z[S_PTS], s2n[S_PTS];
  for (int i = threadIdx.x; i < S_PTS; i += 256) {
    float x = xyz2[i], y = xyz2[S_PTS + i], z = xyz2[2 * S_PTS + i];
    s2x[i] = x; s2y[i] = y; s2z[i] = z; s2n[i] = x * x + y * y + z * z;
  }
  __syncthreads();

  const int n = blockIdx.x * 256 + threadIdx.x;
  const float px = xyz1[n], py = xyz1[N_PTS + n], pz = xyz1[2 * N_PTS + n];
  const float pn = px * px + py * py + pz * pz;

  float d0 = 3.0e38f, d1 = 3.0e38f, d2 = 3.0e38f;
  int   i0 = 0, i1 = 0, i2 = 0;
  for (int s = 0; s < S_PTS; ++s) {
    float d = pn + s2n[s] - 2.0f * (px * s2x[s] + py * s2y[s] + pz * s2z[s]);
    if (d < d2) {
      if (d < d1) {
        d2 = d1; i2 = i1;
        if (d < d0) { d1 = d0; i1 = i0; d0 = d; i0 = s; }
        else        { d1 = d;  i1 = s; }
      } else { d2 = d; i2 = s; }
    }
  }
  const float r0 = 1.0f / (d0 + 1e-8f);
  const float r1 = 1.0f / (d1 + 1e-8f);
  const float r2 = 1.0f / (d2 + 1e-8f);
  const float inv = 1.0f / (r0 + r1 + r2);
  idx[3 * n] = i0; idx[3 * n + 1] = i1; idx[3 * n + 2] = i2;
  wgt[3 * n] = r0 * inv; wgt[3 * n + 1] = r1 * inv; wgt[3 * n + 2] = r2 * inv;
}

// ---------------------------------------------------------------- kernel 2 --
// W0t[j][k] = w0[k][j] (fp16, K-major per output channel), same for W1t.
// gc layout: [0,256)=g0  [256,512)=c0  [512,640)=g1  [640,768)=c1
__global__ __launch_bounds__(256) void prep_kernel(
    const float* __restrict__ w0, const float* __restrict__ b0,
    const float* __restrict__ s0, const float* __restrict__ bb0,
    const float* __restrict__ m0, const float* __restrict__ v0,
    const float* __restrict__ w1, const float* __restrict__ b1,
    const float* __restrict__ s1, const float* __restrict__ bb1,
    const float* __restrict__ m1, const float* __restrict__ v1,
    _Float16* __restrict__ W0t, _Float16* __restrict__ W1t,
    float* __restrict__ gc) {
  int t = blockIdx.x * 256 + threadIdx.x;
  if (t < 256 * 320) {
    int j = t / 320, k = t - j * 320;
    W0t[t] = (_Float16)w0[k * 256 + j];
    return;
  }
  t -= 256 * 320;
  if (t < 128 * 256) {
    int j = t / 256, k = t - j * 256;
    W1t[t] = (_Float16)w1[k * 128 + j];
    return;
  }
  t -= 128 * 256;
  if (t < 256) {
    float g = s0[t] * rsqrtf(v0[t] + 1e-5f);
    gc[t] = g; gc[256 + t] = (b0[t] - m0[t]) * g + bb0[t];
    return;
  }
  t -= 256;
  if (t < 128) {
    float g = s1[t] * rsqrtf(v1[t] + 1e-5f);
    gc[512 + t] = g; gc[640 + t] = (b1[t] - m1[t]) * g + bb1[t];
  }
}

// ---------------------------------------------------------------- kernel 3 --
__global__ __launch_bounds__(256) void fp_fused_kernel(
    const float* __restrict__ points1, const float* __restrict__ points2,
    const int* __restrict__ idx, const float* __restrict__ wgt,
    const _Float16* __restrict__ W0t, const _Float16* __restrict__ W1t,
    const float* __restrict__ gc, float* __restrict__ out) {
  constexpr int MT  = 128;   // dense points per block
  constexpr int LDX = 328;   // halfs; 656B rows (16B aligned, bank-spread)
  constexpr int LDH = 264;   // halfs; 528B rows

  __shared__ _Float16 sX[MT * LDX];   // 82 KB: concat features [r][320]
  __shared__ _Float16 sH[MT * LDH];   // 66 KB: hidden [r][256]
  float* sOut = (float*)sX;           // reuse as [c][132] fp32 staging

  const int tid = threadIdx.x;
  const int n0  = blockIdx.x * MT;

  // ---- build X tile: skip features [0,64) ----
  for (int i = tid; i < MT * 64; i += 256) {
    int d = i >> 7, r = i & 127;
    sX[r * LDX + d] = (_Float16)points1[d * N_PTS + n0 + r];
  }
  // ---- interp features [64,320): two threads per row ----
  {
    const int r = tid >> 1, hs = tid & 1, n = n0 + r;
    const int j0 = idx[3 * n], j1 = idx[3 * n + 1], j2 = idx[3 * n + 2];
    const float a0 = wgt[3 * n], a1 = wgt[3 * n + 1], a2 = wgt[3 * n + 2];
    #pragma unroll 4
    for (int d = hs * 128; d < hs * 128 + 128; ++d) {
      float v = a0 * points2[d * S_PTS + j0] + a1 * points2[d * S_PTS + j1]
              + a2 * points2[d * S_PTS + j2];
      sX[r * LDX + 64 + d] = (_Float16)v;
    }
  }
  __syncthreads();

  const int lane = tid & 31;
  const int wave = tid >> 5;      // 0..7 -> owns a 32-col (GEMM1) / 16-col (GEMM2) slice
  const int h    = lane >> 4;     // half-wave select
  const int l16  = lane & 15;

  // ================= GEMM1: X[128x320] * W0[320x256] =================
  // Wave owns cols [wave*32, wave*32+32): B fragments live in registers,
  // loaded from global exactly once; A streamed from LDS per row-block.
  const float* g0 = gc;       const float* c0 = gc + 256;
  for (int ct2 = 0; ct2 < 2; ++ct2) {
    const int col = (wave * 2 + ct2) * 16 + l16;
    const _Float16* wp = W0t + col * 320;  // K-major column of W0
    FragH bfr[10];                          // 80 VGPRs, resident
    #pragma unroll
    for (int ks = 0; ks < 10; ++ks) {
      // B frag (32x16 f16): lane col = l16; 16 halfs = K[h*16 .. h*16+15]
      bfr[ks].q[0] = *(const f32x4*)(wp + ks * 32 + h * 16);
      bfr[ks].q[1] = *(const f32x4*)(wp + ks * 32 + h * 16 + 8);
    }
    const float gg = g0[col], cc = c0[col];
    for (int rb = 0; rb < 8; ++rb) {
      const _Float16* rp = &sX[(rb * 16 + l16) * LDX];
      v8f acc = {};
      #pragma unroll
      for (int ks = 0; ks < 10; ++ks) {
        // A frag (16x32 f16): lane row = l16; 16 halfs = K[h*16 .. h*16+15]
        FragH af;
        af.q[0] = *(const f32x4*)(rp + ks * 32 + h * 16);
        af.q[1] = *(const f32x4*)(rp + ks * 32 + h * 16 + 8);
        acc = __builtin_amdgcn_wmma_f32_16x16x32_f16(
            false, af.v, false, bfr[ks].v, (short)0, acc, false, false);
      }
      #pragma unroll
      for (int v = 0; v < 8; ++v) {        // D: VGPR v -> row v + h*8
        float y = acc[v] * gg + cc;
        y = y > 0.0f ? y : 0.0f;
        sH[(rb * 16 + v + h * 8) * LDH + col] = (_Float16)y;
      }
    }
  }
  __syncthreads();

  // ================= GEMM2: H[128x256] * W1[256x128] =================
  // Wave owns cols [wave*16, wave*16+16).
  {
    const float* g1 = gc + 512; const float* c1 = gc + 640;
    const int col = wave * 16 + l16;
    const _Float16* wp = W1t + col * 256;
    FragH bfr[8];                           // 64 VGPRs, resident
    #pragma unroll
    for (int ks = 0; ks < 8; ++ks) {
      bfr[ks].q[0] = *(const f32x4*)(wp + ks * 32 + h * 16);
      bfr[ks].q[1] = *(const f32x4*)(wp + ks * 32 + h * 16 + 8);
    }
    const float gg = g1[col], cc = c1[col];
    for (int rb = 0; rb < 8; ++rb) {
      const _Float16* rp = &sH[(rb * 16 + l16) * LDH];
      v8f acc = {};
      #pragma unroll
      for (int ks = 0; ks < 8; ++ks) {
        FragH af;
        af.q[0] = *(const f32x4*)(rp + ks * 32 + h * 16);
        af.q[1] = *(const f32x4*)(rp + ks * 32 + h * 16 + 8);
        acc = __builtin_amdgcn_wmma_f32_16x16x32_f16(
            false, af.v, false, bfr[ks].v, (short)0, acc, false, false);
      }
      #pragma unroll
      for (int v = 0; v < 8; ++v) {
        float y = acc[v] * gg + cc;
        y = y > 0.0f ? y : 0.0f;
        sOut[col * 132 + rb * 16 + v + h * 8] = y;  // channel-major staging
      }
    }
  }
  __syncthreads();

  // ---- coalesced store: out[c][n0 + r], float4 per thread ----
  for (int i = tid; i < 128 * 32; i += 256) {
    int c = i >> 5, rq = i & 31;
    f32x4 v = *(const f32x4*)(sOut + c * 132 + rq * 4);
    *(f32x4*)(out + (size_t)c * N_PTS + n0 + rq * 4) = v;
  }
}

// ------------------------------------------------------------------ launch --
extern "C" void kernel_launch(void* const* d_in, const int* in_sizes, int n_in,
                              void* d_out, int out_size, void* d_ws, size_t ws_size,
                              hipStream_t stream) {
  const float* xyz1    = (const float*)d_in[0];
  const float* xyz2    = (const float*)d_in[1];
  const float* points1 = (const float*)d_in[2];
  const float* points2 = (const float*)d_in[3];
  const float* w0  = (const float*)d_in[4];
  const float* b0  = (const float*)d_in[5];
  const float* s0  = (const float*)d_in[6];
  const float* bb0 = (const float*)d_in[7];
  const float* m0  = (const float*)d_in[8];
  const float* v0  = (const float*)d_in[9];
  const float* w1  = (const float*)d_in[10];
  const float* b1  = (const float*)d_in[11];
  const float* s1  = (const float*)d_in[12];
  const float* bb1 = (const float*)d_in[13];
  const float* m1  = (const float*)d_in[14];
  const float* v1  = (const float*)d_in[15];

  char* ws = (char*)d_ws;                      // all offsets 16B aligned
  int*      idx = (int*)ws;                    // 3*N ints     (768 KB)
  float*    wgt = (float*)(ws + 786432);       // 3*N floats   (768 KB)
  _Float16* W0t = (_Float16*)(ws + 1572864);   // 256*320 f16  (160 KB)
  _Float16* W1t = (_Float16*)(ws + 1736704);   // 128*256 f16  ( 64 KB)
  float*    gc  = (float*)(ws + 1802240);      // 768 floats

  top3_kernel<<<N_PTS / 256, 256, 0, stream>>>(xyz1, xyz2, idx, wgt);

  const int prep_work = 256 * 320 + 128 * 256 + 256 + 128;
  prep_kernel<<<(prep_work + 255) / 256, 256, 0, stream>>>(
      w0, b0, s0, bb0, m0, v0, w1, b1, s1, bb1, m1, v1, W0t, W1t, gc);

  fp_fused_kernel<<<N_PTS / 128, 256, 0, stream>>>(
      points1, points2, idx, wgt, W0t, W1t, gc, (float*)d_out);
}
